// DotProductLabelling_3599182594335
// MI455X (gfx1250) — 16-bit and 8-bit matrix operand order repaired and checked by reading, compiled, not run on hardware
//
#include <hip/hip_runtime.h>
#include <hip/hip_bf16.h>

#define NN     2048
#define EE     262144
#define INCH   256
#define HID    256
#define D1PAD  288

typedef __attribute__((ext_vector_type(8)))  int    v8i;
typedef __attribute__((ext_vector_type(8)))  float  v8f;
typedef __attribute__((ext_vector_type(16))) __bf16 v16bf;

#define GLOBAL_AS __attribute__((address_space(1)))
#define LDS_AS    __attribute__((address_space(3)))

typedef int v4i_gv __attribute__((vector_size(16)));

#if __has_builtin(__builtin_amdgcn_global_load_async_to_lds_b128)
#define HAVE_ASYNC_LDS 1
#endif

__device__ __forceinline__ void async_cp16(const void* g, void* l) {
#if defined(HAVE_ASYNC_LDS)
    __builtin_amdgcn_global_load_async_to_lds_b128(
        (GLOBAL_AS v4i_gv*)(g), (LDS_AS v4i_gv*)(l), 0, 0);
#else
    *(uint4*)l = *(const uint4*)g;
#endif
}
__device__ __forceinline__ void wait_async_lds() {
#if __has_builtin(__builtin_amdgcn_s_wait_asynccnt)
    __builtin_amdgcn_s_wait_asynccnt(0);
#elif defined(HAVE_ASYNC_LDS)
    asm volatile("s_wait_asynccnt 0" ::: "memory");
#endif
}

__device__ __forceinline__ unsigned short f2bf(float f) {
    unsigned int u = __float_as_uint(f);
    unsigned int r = u + 0x7FFFu + ((u >> 16) & 1u);
    return (unsigned short)(r >> 16);
}
__device__ __forceinline__ float bf2f(unsigned short h) {
    return __uint_as_float(((unsigned int)h) << 16);
}

__device__ __forceinline__ v8i ld_afrag_u8(const unsigned char* base, int row,
                                           int ld, int kbase, int hi) {
    const unsigned char* p = base + (size_t)row * ld + kbase + hi * 16;
    uint2 a = *(const uint2*)(p);
    uint2 b = *(const uint2*)(p + 8);
    uint2 c = *(const uint2*)(p + 32);
    uint2 d = *(const uint2*)(p + 40);
    v8i f;
    f[0]=(int)a.x; f[1]=(int)a.y; f[2]=(int)b.x; f[3]=(int)b.y;
    f[4]=(int)c.x; f[5]=(int)c.y; f[6]=(int)d.x; f[7]=(int)d.y;
    return f;
}
__device__ __forceinline__ v8i ld_bfrag_u8(const unsigned char* base, int col,
                                           int ld, int kbase, int hi) {
    const unsigned char* p = base + (size_t)col * ld + kbase + hi * 16;
    uint4 a = *(const uint4*)(p);
    uint4 b = *(const uint4*)(p + 32);
    v8i f;
    f[0]=(int)a.x; f[1]=(int)a.y; f[2]=(int)a.z; f[3]=(int)a.w;
    f[4]=(int)b.x; f[5]=(int)b.y; f[6]=(int)b.z; f[7]=(int)b.w;
    return f;
}
__device__ __forceinline__ v16bf ld_afrag_bf16(const unsigned short* base, int row,
                                               int ld, int kbase, int hi) {
    const unsigned short* p = base + (size_t)row * ld + kbase + hi * 16;
    struct U { uint4 a, b; } r;
    r.a = *(const uint4*)(p);
    r.b = *(const uint4*)(p + 8);
    return __builtin_bit_cast(v16bf, r);
}
__device__ __forceinline__ v16bf ld_bfrag_bf16(const unsigned short* base, int col,
                                               int ld, int kbase, int hi) {
    const unsigned short* p = base + (size_t)col * ld + kbase + hi * 16;
    struct U { uint4 a, b; } r;
    r.a = *(const uint4*)(p);
    r.b = *(const uint4*)(p + 8);
    return __builtin_bit_cast(v16bf, r);
}

__global__ __launch_bounds__(256) void pack_adj_kernel(const float* __restrict__ A,
                                                       unsigned char* __restrict__ Au8) {
    int idx = blockIdx.x * 256 + threadIdx.x;
    float4 f = ((const float4*)A)[idx];
    uchar4 b;
    b.x = (unsigned char)f.x; b.y = (unsigned char)f.y;
    b.z = (unsigned char)f.z; b.w = (unsigned char)f.w;
    ((uchar4*)Au8)[idx] = b;
}

#define GBM 128
#define GBN 64
__global__ __launch_bounds__(256) void gemm_iu8_kernel(const unsigned char* __restrict__ A,
                                                       const unsigned char* __restrict__ Bs,
                                                       int* __restrict__ C) {
    __shared__ __attribute__((aligned(16))) unsigned char bufA[2][GBM * 64];
    __shared__ __attribute__((aligned(16))) unsigned char bufB[2][GBN * 64];

    const int tid  = threadIdx.x;
    const int row0 = (blockIdx.x >> 5) * GBM;
    const int col0 = (blockIdx.x & 31) * GBN;
    const int wid  = tid >> 5;
    const int lane = tid & 31;
    const int sub  = lane & 15;
    const int hi   = lane >> 4;

    auto stage = [&](int buf, int k0) {
        #pragma unroll
        for (int h = 0; h < 2; ++h) {
            int i   = tid + h * 256;
            int row = i >> 2, ch = i & 3;
            async_cp16(A + (size_t)(row0 + row) * NN + k0 + ch * 16,
                       &bufA[buf][row * 64 + ch * 16]);
        }
        {
            int col = tid >> 2, ch = tid & 3;
            async_cp16(Bs + (size_t)(col0 + col) * NN + k0 + ch * 16,
                       &bufB[buf][col * 64 + ch * 16]);
        }
    };

    v8i z = {0,0,0,0,0,0,0,0};
    v8i acc0 = z, acc1 = z, acc2 = z, acc3 = z;

    stage(0, 0);
    for (int k0 = 0; k0 < NN; k0 += 64) {
        int cur = (k0 >> 6) & 1;
        wait_async_lds();
        __syncthreads();
        if (k0 + 64 < NN) stage(cur ^ 1, k0 + 64);

        v8i a = ld_afrag_u8(&bufA[cur][0], wid * 16 + sub, 64, 0, hi);
        v8i b0 = ld_bfrag_u8(&bufB[cur][0], 0 * 16 + sub, 64, 0, hi);
        acc0 = __builtin_amdgcn_wmma_i32_16x16x64_iu8(false, a, false, b0, acc0, false, false);
        v8i b1 = ld_bfrag_u8(&bufB[cur][0], 1 * 16 + sub, 64, 0, hi);
        acc1 = __builtin_amdgcn_wmma_i32_16x16x64_iu8(false, a, false, b1, acc1, false, false);
        v8i b2 = ld_bfrag_u8(&bufB[cur][0], 2 * 16 + sub, 64, 0, hi);
        acc2 = __builtin_amdgcn_wmma_i32_16x16x64_iu8(false, a, false, b2, acc2, false, false);
        v8i b3 = ld_bfrag_u8(&bufB[cur][0], 3 * 16 + sub, 64, 0, hi);
        acc3 = __builtin_amdgcn_wmma_i32_16x16x64_iu8(false, a, false, b3, acc3, false, false);
    }

    const int mbase = row0 + wid * 16 + hi * 8;
    #pragma unroll
    for (int r = 0; r < 8; ++r) {
        size_t rowoff = (size_t)(mbase + r) * NN + col0 + sub;
        C[rowoff +  0] = acc0[r];
        C[rowoff + 16] = acc1[r];
        C[rowoff + 32] = acc2[r];
        C[rowoff + 48] = acc3[r];
    }
}

__global__ __launch_bounds__(256) void make_b2_kernel(const int* __restrict__ A2c,
                                                      const unsigned char* __restrict__ Au8,
                                                      unsigned char* __restrict__ B2u8) {
    int idx = blockIdx.x * 256 + threadIdx.x;
    int i = idx >> 11, j = idx & (NN - 1);
    B2u8[idx] = (A2c[idx] > 0 && Au8[idx] == 0 && i != j) ? 1 : 0;
}

__global__ __launch_bounds__(256) void conv_weights_kernel(const float* __restrict__ W1,
                                                           const float* __restrict__ W2,
                                                           const float* __restrict__ W3,
                                                           unsigned short* __restrict__ W1p,
                                                           unsigned short* __restrict__ W2b,
                                                           unsigned short* __restrict__ W3b) {
    int idx = blockIdx.x * 256 + threadIdx.x;
    const int n1 = HID * D1PAD;
    const int n2 = HID * HID;
    if (idx < n1) {
        int n = idx / D1PAD, k = idx % D1PAD;
        float v = (k < 261) ? W1[n * 261 + k] : 0.0f;
        W1p[n * D1PAD + k] = f2bf(v);
    } else if (idx < n1 + n2) {
        int t = idx - n1;
        W2b[t] = f2bf(W2[t]);
    } else if (idx < n1 + n2 + HID) {
        int k = idx - n1 - n2;
        W3b[k] = f2bf(W3[k]);
    }
}

#define EPB 64
__global__ __launch_bounds__(256) void edge_mlp_kernel(const float* __restrict__ x,
                                                       const int* __restrict__ edges,
                                                       const unsigned char* __restrict__ Au8,
                                                       const int* __restrict__ A2c,
                                                       const int* __restrict__ AB,
                                                       const int* __restrict__ B2B2,
                                                       const unsigned short* __restrict__ W1p,
                                                       const unsigned short* __restrict__ W2b,
                                                       const unsigned short* __restrict__ W3b,
                                                       const float* __restrict__ b1,
                                                       const float* __restrict__ b2,
                                                       const float* __restrict__ b3,
                                                       float* __restrict__ out) {
    __shared__ __attribute__((aligned(16))) unsigned short Apan[EPB][D1PAD];
    __shared__ __attribute__((aligned(16))) unsigned short Hbuf[EPB][HID];
    __shared__ int   uv[EPB][2];
    __shared__ float red[EPB][4];

    const int tid = threadIdx.x;
    const int e0  = blockIdx.x * EPB;

    if (tid < 2 * EPB) {
        int which = tid & 1, i = tid >> 1;
        uv[i][which] = edges[which * EE + e0 + i];
    }
    __syncthreads();

    for (int t = tid; t < EPB * INCH; t += 256) {
        int i = t >> 8, c = t & 255;
        int u = uv[i][0], v = uv[i][1];
        float p = x[(size_t)u * INCH + c] * x[(size_t)v * INCH + c];
        Apan[i][c] = f2bf(p);
    }
    for (int t = tid; t < EPB * 32; t += 256) {
        int i = t >> 5, c = 256 + (t & 31);
        int u = uv[i][0], v = uv[i][1];
        float f = 0.0f;
        if      (c == 256) f = (float)Au8 [(size_t)u * NN + v];
        else if (c == 257) f = (float)A2c [(size_t)u * NN + v];
        else if (c == 258) f = (float)AB  [(size_t)u * NN + v];
        else if (c == 259) f = (float)AB  [(size_t)v * NN + u];
        else if (c == 260) f = (float)B2B2[(size_t)u * NN + v];
        Apan[i][c] = f2bf(f);
    }
    __syncthreads();

    const int wid = tid >> 5, lane = tid & 31;
    const int sub = lane & 15, hi = lane >> 4;
    v8f zf = {0,0,0,0,0,0,0,0};

    {
        v8f acc[4][2];
        #pragma unroll
        for (int mt = 0; mt < 4; ++mt) { acc[mt][0] = zf; acc[mt][1] = zf; }
        for (int k0 = 0; k0 < D1PAD; k0 += 32) {
            v16bf bf0 = ld_bfrag_bf16(W1p, wid * 32 + 0 * 16 + sub, D1PAD, k0, hi);
            v16bf bf1 = ld_bfrag_bf16(W1p, wid * 32 + 1 * 16 + sub, D1PAD, k0, hi);
            #pragma unroll
            for (int mt = 0; mt < 4; ++mt) {
                v16bf a = ld_afrag_bf16(&Apan[0][0], mt * 16 + sub, D1PAD, k0, hi);
                acc[mt][0] = __builtin_amdgcn_wmma_f32_16x16x32_bf16(
                    false, a, false, bf0, (short)0, acc[mt][0], false, false);
                acc[mt][1] = __builtin_amdgcn_wmma_f32_16x16x32_bf16(
                    false, a, false, bf1, (short)0, acc[mt][1], false, false);
            }
        }
        #pragma unroll
        for (int j = 0; j < 2; ++j) {
            int n = wid * 32 + j * 16 + sub;
            float bias = b1[n];
            #pragma unroll
            for (int mt = 0; mt < 4; ++mt)
                #pragma unroll
                for (int r = 0; r < 8; ++r)
                    Hbuf[mt * 16 + r + hi * 8][n] = f2bf(fmaxf(acc[mt][j][r] + bias, 0.0f));
        }
    }
    __syncthreads();

    {
        v8f acc[4][2];
        #pragma unroll
        for (int mt = 0; mt < 4; ++mt) { acc[mt][0] = zf; acc[mt][1] = zf; }
        for (int k0 = 0; k0 < HID; k0 += 32) {
            v16bf bf0 = ld_bfrag_bf16(W2b, wid * 32 + 0 * 16 + sub, HID, k0, hi);
            v16bf bf1 = ld_bfrag_bf16(W2b, wid * 32 + 1 * 16 + sub, HID, k0, hi);
            #pragma unroll
            for (int mt = 0; mt < 4; ++mt) {
                v16bf a = ld_afrag_bf16(&Hbuf[0][0], mt * 16 + sub, HID, k0, hi);
                acc[mt][0] = __builtin_amdgcn_wmma_f32_16x16x32_bf16(
                    false, a, false, bf0, (short)0, acc[mt][0], false, false);
                acc[mt][1] = __builtin_amdgcn_wmma_f32_16x16x32_bf16(
                    false, a, false, bf1, (short)0, acc[mt][1], false, false);
            }
        }
        #pragma unroll
        for (int j = 0; j < 2; ++j) {
            int n = wid * 32 + j * 16 + sub;
            float bias = b2[n];
            #pragma unroll
            for (int mt = 0; mt < 4; ++mt)
                #pragma unroll
                for (int r = 0; r < 8; ++r)
                    Apan[mt * 16 + r + hi * 8][n] = f2bf(fmaxf(acc[mt][j][r] + bias, 0.0f));
        }
    }
    __syncthreads();

    {
        int i = tid >> 2, part = tid & 3;
        int kb = part * 64;
        float s = 0.0f;
        #pragma unroll
        for (int k = 0; k < 64; ++k)
            s += bf2f(Apan[i][kb + k]) * bf2f(W3b[kb + k]);
        red[i][part] = s;
    }
    __syncthreads();
    if (tid < EPB) {
        float t = b3[0] + red[tid][0] + red[tid][1] + red[tid][2] + red[tid][3];
        out[e0 + tid] = t;
    }
}

#define OFF_AU8   ((size_t)0)
#define OFF_B2U8  ((size_t)(NN * NN))
#define OFF_A2C   ((size_t)(2ull * NN * NN))
#define OFF_AB    (OFF_A2C  + (size_t)NN * NN * 4)
#define OFF_B2B2  (OFF_AB   + (size_t)NN * NN * 4)
#define OFF_W1P   (OFF_B2B2 + (size_t)NN * NN * 4)
#define OFF_W2B   (OFF_W1P  + (size_t)HID * D1PAD * 2)
#define OFF_W3B   (OFF_W2B  + (size_t)HID * HID * 2)

extern "C" void kernel_launch(void* const* d_in, const int* in_sizes, int n_in,
                              void* d_out, int out_size, void* d_ws, size_t ws_size,
                              hipStream_t stream) {
    const float* x    = (const float*)d_in[0];
    const float* adj  = (const float*)d_in[1];
    const int*   edges= (const int*)  d_in[2];
    const float* W1   = (const float*)d_in[3];
    const float* b1   = (const float*)d_in[4];
    const float* W2   = (const float*)d_in[5];
    const float* b2   = (const float*)d_in[6];
    const float* W3   = (const float*)d_in[7];
    const float* b3   = (const float*)d_in[8];
    float* out = (float*)d_out;

    char* ws = (char*)d_ws;
    unsigned char*  Au8  = (unsigned char*) (ws + OFF_AU8);
    unsigned char*  B2u8 = (unsigned char*) (ws + OFF_B2U8);
    int*            A2c  = (int*)           (ws + OFF_A2C);
    int*            AB   = (int*)           (ws + OFF_AB);
    int*            B2B2 = (int*)           (ws + OFF_B2B2);
    unsigned short* W1p  = (unsigned short*)(ws + OFF_W1P);
    unsigned short* W2b  = (unsigned short*)(ws + OFF_W2B);
    unsigned short* W3b  = (unsigned short*)(ws + OFF_W3B);

    pack_adj_kernel<<<(NN * NN / 4) / 256, 256, 0, stream>>>(adj, Au8);
    {
        int total = HID * D1PAD + HID * HID + HID;
        conv_weights_kernel<<<(total + 255) / 256, 256, 0, stream>>>(W1, W2, W3, W1p, W2b, W3b);
    }
    gemm_iu8_kernel<<<512, 256, 0, stream>>>(Au8, Au8, A2c);
    make_b2_kernel<<<(NN * NN) / 256, 256, 0, stream>>>(A2c, Au8, B2u8);
    gemm_iu8_kernel<<<512, 256, 0, stream>>>(Au8, B2u8, AB);
    gemm_iu8_kernel<<<512, 256, 0, stream>>>(B2u8, B2u8, B2B2);
    edge_mlp_kernel<<<EE / EPB, 256, 0, stream>>>(x, edges, Au8, A2c, AB, B2B2,
                                                  W1p, W2b, W3b, b1, b2, b3, out);
}
